// ScaledDotProductAttention_65584150609972
// MI455X (gfx1250) — matrix operand order repaired and checked by reading, not compiled, not run on hardware
//
#include <hip/hip_runtime.h>

// ---------------------------------------------------------------------------
// Scaled dot-product attention forward for MI455X (gfx1250, wave32, WMMA).
//   B=2, H=16, S=2048, D=64. fp32 in/out, f16 WMMA math with fp32 accumulate.
// One workgroup (8 waves / 256 threads) = one (b,h) head x 16-query tile.
// Phase 1: S = Q K^T / 8 via v_wmma_f32_16x16x32_f16, masked fill=1e-9,
//          stored f16 into a 16x2048 LDS tile (possible only with CDNA5's
//          320KB WGP LDS).
// Softmax:  row max/sum via b128 LDS reads + small LDS reductions; attn
//          written to HBM with coalesced b128 stores (the 536MB stream that
//          dominates the roofline).
// Phase 2: context = P V via WMMA; V transposed through per-wave LDS tiles
//          (pad-staggered, conflict-free ds_load_b128); cross-wave reduce
//          with ds_add_f32.
// ---------------------------------------------------------------------------

#define S_LEN   2048
#define DK      64
#define NWAVES  8
#define KPW     (S_LEN / NWAVES)   // 256 keys per wave
#define PST     2056               // padded Ps row stride (f16 elems): banks rotate
#define VST     40                 // padded Vt row stride (f16 elems): 20 dw, gcd(20,64)=4

typedef __attribute__((ext_vector_type(16))) _Float16 v16h;
typedef __attribute__((ext_vector_type(8)))  _Float16 v8h;
typedef __attribute__((ext_vector_type(8)))  float    v8f;
typedef __attribute__((ext_vector_type(4)))  float    f32x4;

static __device__ inline v16h cat8(v8h lo, v8h hi) {
  return __builtin_shufflevector(lo, hi, 0,1,2,3,4,5,6,7,8,9,10,11,12,13,14,15);
}

__global__ __launch_bounds__(256, 1)
void attn_fwd_wmma(const float* __restrict__ Q, const float* __restrict__ K,
                   const float* __restrict__ V,
                   const unsigned char* __restrict__ mask,   // [B,H,S] bool (1B)
                   float* __restrict__ outCtx,               // [B,H,S,64]
                   float* __restrict__ outAttn)              // [B,H,S,S]
{
  __shared__ _Float16 Qs[16 * DK];              //   2 KB
  __shared__ _Float16 Ps[16 * PST];             //  64 KB scores -> probs (f16)
  __shared__ _Float16 Vt[NWAVES][DK * VST];     //  40 KB transposed V chunks
  __shared__ float    red[16][16];              //   1 KB row reductions
  __shared__ float    ctx[16 * DK];             //   4 KB context accumulator
  __shared__ float    mrow[16];                 // query-row mask (0/1)

  const int tid  = threadIdx.x;
  const int lane = tid & 31;                    // wave32
  const int wave = tid >> 5;
  const int m    = lane & 15;                   // M row / N col index within tile
  const int seg  = lane >> 4;                   // lane half selector
  const int bh   = blockIdx.y;                  // 0..31  (b*16+h)
  const int qb   = blockIdx.x * 16;             // query tile base

  const size_t headOff = (size_t)bh * S_LEN * DK;
  const float* Qh = Q + headOff + (size_t)qb * DK;
  const float* Kh = K + headOff;
  const float* Vh = V + headOff;
  const unsigned char* mh = mask + (size_t)bh * S_LEN;

  // ---- stage Q tile (f32 -> f16), row mask, zero context accumulator ----
  for (int e = tid; e < 16 * DK; e += 256) Qs[e] = (_Float16)Qh[e];
  for (int e = tid; e < 16 * DK; e += 256) ctx[e] = 0.0f;
  if (tid < 16) mrow[tid] = (mh[qb + tid] != 0) ? 1.0f : 0.0f;
  __syncthreads();

  // ================= Phase 1: scores = Q K^T * 0.125, masked ==============
  // A operand layout (16x32 f16): lane holds row m, 16 contiguous K from seg*16
  const v8h* qrow = (const v8h*)(Qs + m * DK);
  const v16h a0 = cat8(qrow[0 + 2 * seg], qrow[1 + 2 * seg]);   // d =  0..31
  const v16h a1 = cat8(qrow[4 + 2 * seg], qrow[5 + 2 * seg]);   // d = 32..63

  float mq[8];
  #pragma unroll
  for (int j = 0; j < 8; ++j) mq[j] = mrow[seg * 8 + j];

  const int keyw = wave * KPW;
  for (int kt = 0; kt < KPW / 16; ++kt) {
    const int keyn = keyw + kt * 16 + m;        // this lane's key column (B: n=m)
    const float* kp = Kh + (size_t)keyn * DK + seg * 16;
    // B operand (32x16 f16): lane holds column keyn, 16 contiguous d values
    v16h b0, b1;
    {
      f32x4 f0 = ((const f32x4*)kp)[0];
      f32x4 f1 = ((const f32x4*)kp)[1];
      f32x4 f2 = ((const f32x4*)kp)[2];
      f32x4 f3 = ((const f32x4*)kp)[3];
      f32x4 g0 = ((const f32x4*)(kp + 32))[0];
      f32x4 g1 = ((const f32x4*)(kp + 32))[1];
      f32x4 g2 = ((const f32x4*)(kp + 32))[2];
      f32x4 g3 = ((const f32x4*)(kp + 32))[3];
      #pragma unroll
      for (int j = 0; j < 4; ++j) {
        b0[j]      = (_Float16)f0[j];  b0[4 + j]  = (_Float16)f1[j];
        b0[8 + j]  = (_Float16)f2[j];  b0[12 + j] = (_Float16)f3[j];
        b1[j]      = (_Float16)g0[j];  b1[4 + j]  = (_Float16)g1[j];
        b1[8 + j]  = (_Float16)g2[j];  b1[12 + j] = (_Float16)g3[j];
      }
    }
    v8f acc = {0.f, 0.f, 0.f, 0.f, 0.f, 0.f, 0.f, 0.f};
    acc = __builtin_amdgcn_wmma_f32_16x16x32_f16(false, a0, false, b0,
                                                 (short)0, acc, false, false);
    acc = __builtin_amdgcn_wmma_f32_16x16x32_f16(false, a1, false, b1,
                                                 (short)0, acc, false, false);
    // C layout: vgpr j -> row (j + seg*8), col m (== keyn here)
    const bool mk = (mh[keyn] != 0);
    #pragma unroll
    for (int j = 0; j < 8; ++j) {
      const int r = j + seg * 8;
      const float s = acc[j] * 0.125f;                 // 1/sqrt(64)
      const float val = (mk && (mq[j] != 0.0f)) ? s : 1e-9f;
      Ps[r * PST + keyn] = (_Float16)val;
    }
  }
  __syncthreads();

  // ======================= Softmax over each row ==========================
  {
    const int r    = tid >> 4;            // 16 threads per row
    const int jcol = tid & 15;
    _Float16* prow = Ps + r * PST;
    // pass 1: row max (b128 reads, conflict-free: banks = jcol*4)
    float M = -3.0e38f;
    #pragma unroll 4
    for (int i = 0; i < 16; ++i) {
      const v8h g = ((const v8h*)prow)[jcol + 16 * i];
      #pragma unroll
      for (int j = 0; j < 8; ++j) M = fmaxf(M, (float)g[j]);
    }
    red[r][jcol] = M;
    __syncthreads();
    #pragma unroll
    for (int j = 0; j < 16; ++j) M = fmaxf(M, red[r][j]);
    __syncthreads();
    // pass 2: exp(s - M), store back, accumulate sum
    float Lsum = 0.0f;
    #pragma unroll 4
    for (int i = 0; i < 16; ++i) {
      const v8h g = ((const v8h*)prow)[jcol + 16 * i];
      v8h e;
      #pragma unroll
      for (int j = 0; j < 8; ++j) {
        const float ev = __expf((float)g[j] - M);
        Lsum += ev;
        e[j] = (_Float16)ev;
      }
      ((v8h*)prow)[jcol + 16 * i] = e;
    }
    red[r][jcol] = Lsum;
    __syncthreads();
    float Ltot = 0.0f;
    #pragma unroll
    for (int j = 0; j < 16; ++j) Ltot += red[r][j];
    const float inv = 1.0f / Ltot;
    // pass 3: p = e * inv -> Ps (f16 for phase 2) and attn out (coalesced b128)
    float* arow = outAttn + ((size_t)bh * S_LEN + qb + r) * S_LEN;
    #pragma unroll 4
    for (int i = 0; i < 16; ++i) {
      const int grp = jcol + 16 * i;
      const v8h g = ((const v8h*)prow)[grp];
      v8h pq; f32x4 o0, o1;
      #pragma unroll
      for (int j = 0; j < 8; ++j) {
        const float p = (float)g[j] * inv;
        pq[j] = (_Float16)p;
        if (j < 4) o0[j] = p; else o1[j - 4] = p;
      }
      ((v8h*)prow)[grp] = pq;
      ((f32x4*)arow)[grp * 2]     = o0;
      ((f32x4*)arow)[grp * 2 + 1] = o1;
    }
  }
  __syncthreads();

  // ================= Phase 2: context = P V (per-wave key slice) ==========
  v8f cacc[4];
  #pragma unroll
  for (int nt = 0; nt < 4; ++nt)
    cacc[nt] = (v8f){0.f, 0.f, 0.f, 0.f, 0.f, 0.f, 0.f, 0.f};

  _Float16* vtw = Vt[wave];
  for (int ch = 0; ch < KPW / 32; ++ch) {
    const int kb = keyw + ch * 32;
    // Transpose V[kb..kb+31][0..63] into Vt[d][k] (lane owns rows d=lane,lane+32).
    // Global reads: at fixed k, 32 lanes read 128B contiguous (coalesced).
    _Float16 t0[32], t1[32];
    #pragma unroll
    for (int k = 0; k < 32; ++k) {
      const float* vp = Vh + (size_t)(kb + k) * DK + lane;
      t0[k] = (_Float16)vp[0];
      t1[k] = (_Float16)vp[32];
    }
    v8h* r0 = (v8h*)(vtw + lane * VST);
    v8h* r1 = (v8h*)(vtw + (lane + 32) * VST);
    #pragma unroll
    for (int qd = 0; qd < 4; ++qd) {
      v8h s0, s1;
      #pragma unroll
      for (int j = 0; j < 8; ++j) { s0[j] = t0[qd * 8 + j]; s1[j] = t1[qd * 8 + j]; }
      r0[qd] = s0;
      r1[qd] = s1;
    }
    // A operand = probabilities: row m, 16 contiguous keys from kb+seg*16
    const _Float16* pr = Ps + m * PST;
    const v16h pa = cat8(*(const v8h*)(pr + kb + seg * 16),
                         *(const v8h*)(pr + kb + seg * 16 + 8));
    // B operand = Vt rows: lane holds column d = nt*16+m, 16 contiguous keys
    #pragma unroll
    for (int nt = 0; nt < 4; ++nt) {
      const v8h* vr = (const v8h*)(vtw + (nt * 16 + m) * VST + seg * 16);
      const v16h vb = cat8(vr[0], vr[1]);
      cacc[nt] = __builtin_amdgcn_wmma_f32_16x16x32_f16(false, pa, false, vb,
                                                        (short)0, cacc[nt],
                                                        false, false);
    }
  }

  // Cross-wave reduction of partial 16x64 contexts via ds_add_f32.
  #pragma unroll
  for (int nt = 0; nt < 4; ++nt) {
    #pragma unroll
    for (int j = 0; j < 8; ++j) {
      const int r = j + seg * 8;
      const int d = nt * 16 + m;
      atomicAdd(&ctx[r * DK + d], cacc[nt][j]);
    }
  }
  __syncthreads();

  float* oc = outCtx + ((size_t)bh * S_LEN + qb) * DK;
  for (int e = tid; e < 16 * DK; e += 256) oc[e] = ctx[e];
}

extern "C" void kernel_launch(void* const* d_in, const int* in_sizes, int n_in,
                              void* d_out, int out_size, void* d_ws, size_t ws_size,
                              hipStream_t stream) {
  (void)in_sizes; (void)n_in; (void)out_size; (void)d_ws; (void)ws_size;
  const float* Q = (const float*)d_in[0];
  const float* K = (const float*)d_in[1];
  const float* V = (const float*)d_in[2];
  const unsigned char* mask = (const unsigned char*)d_in[3];  // jnp.bool_: 1B/elem

  float* outCtx  = (float*)d_out;                              // [2,16,2048,64]
  float* outAttn = outCtx + (size_t)2 * 16 * 2048 * 64;        // [2,16,2048,2048]

  dim3 grid(S_LEN / 16, 2 * 16);   // 128 query tiles x 32 heads
  attn_fwd_wmma<<<grid, 256, 0, stream>>>(Q, K, V, mask, outCtx, outAttn);
}
